// TransformerBlock_77103252898332
// MI455X (gfx1250) — matrix operand order repaired and checked by reading, not compiled, not run on hardware
//
#include <hip/hip_runtime.h>
#include <cstddef>

// ---------------------------------------------------------------------------
// Types for CDNA5 WMMA (wave32). D = A(16x32 bf16) * B(32x16 bf16) + C(16x16 f32)
// ---------------------------------------------------------------------------
typedef __attribute__((ext_vector_type(16))) __bf16 v16bf;
typedef __attribute__((ext_vector_type(8)))  __bf16 v8bf;
typedef __attribute__((ext_vector_type(8)))  float  v8f;

typedef int v4i __attribute__((vector_size(16)));
typedef __attribute__((address_space(1))) v4i as1_v4i;
typedef __attribute__((address_space(3))) v4i as3_v4i;

union FragBF { v16bf v; v8bf h[2]; };

__device__ __forceinline__ v8f vzero8() {
  v8f z;
#pragma unroll
  for (int e = 0; e < 8; ++e) z[e] = 0.f;
  return z;
}

// ---- CDNA5 async global->LDS copy (ASYNCcnt path), guarded fallback --------
#if defined(__gfx1250__) && __has_builtin(__builtin_amdgcn_global_load_async_to_lds_b128)
#define HAVE_ASYNC 1
#else
#define HAVE_ASYNC 0
#endif

__device__ __forceinline__ void async_cp16(const __bf16* g, __bf16* l) {
#if HAVE_ASYNC
  __builtin_amdgcn_global_load_async_to_lds_b128(
      (as1_v4i*)(unsigned long long)g,
      (as3_v4i*)(unsigned)(unsigned long long)l, 0, 0);
#else
  *(v8bf*)l = *(const v8bf*)g;
#endif
}

#if defined(__gfx1250__) && __has_builtin(__builtin_amdgcn_s_wait_asynccnt)
#define WAIT_ASYNCCNT(n) __builtin_amdgcn_s_wait_asynccnt(n)
#elif defined(__gfx1250__)
#define WAIT_ASYNCCNT(n) asm volatile("s_wait_asynccnt %0" ::"i"(n) : "memory")
#else
#define WAIT_ASYNCCNT(n)
#endif

// A-matrix 16x32 bf16 fragment (ISA 7.12.2):
//   lanes 0-15 : row = lane,    elems 0-7 -> K 0-7,  elems 8-15 -> K 16-23
//   lanes 16-31: row = lane-16, elems 0-7 -> K 8-15, elems 8-15 -> K 24-31
template <typename P>
__device__ __forceinline__ v16bf load_a_frag(P base, int stride, int row0, int kbase, int lane) {
  const int r = lane & 15, hf = lane >> 4;
  P p = base + (size_t)(row0 + r) * stride + kbase + hf * 8;
  FragBF u;
  u.h[0] = *(const v8bf*)p;
  u.h[1] = *(const v8bf*)(p + 16);
  return u.v;
}

// B-matrix 32x16 bf16 fragment: lane holds one column (N = lane&15),
// elems 0-7 -> K half*8+0..7, elems 8-15 -> K 16+half*8+0..7. base is row-major [N][K].
template <typename P>
__device__ __forceinline__ v16bf load_b_frag(P base, int stride, int col0, int kbase, int lane) {
  const int r = lane & 15, hf = lane >> 4;
  P p = base + (size_t)(col0 + r) * stride + kbase + hf * 8;
  FragBF u;
  u.h[0] = *(const v8bf*)p;
  u.h[1] = *(const v8bf*)(p + 16);
  return u.v;
}

// ---------------------------------------------------------------------------
// fp32 -> bf16 conversion
// ---------------------------------------------------------------------------
__global__ void cvt_f32_bf16(const float* __restrict__ src, __bf16* __restrict__ dst, int n) {
  int i = blockIdx.x * 256 + threadIdx.x;
  if (i < n) dst[i] = (__bf16)src[i];
}

// ---------------------------------------------------------------------------
// Build concatenated KAN weight: row o of [O][IN*9]:
//   cols [0,IN)       = base_w[o][i]
//   cols IN + i*8 + c = spline_w[o][i][c] * scaler[o][i]
// ---------------------------------------------------------------------------
__global__ void build_wcat(const float* __restrict__ base_w, const float* __restrict__ spline_w,
                           const float* __restrict__ scaler, __bf16* __restrict__ wcat,
                           int total, int IN) {
  int idx = blockIdx.x * 256 + threadIdx.x;
  if (idx >= total) return;
  int o = idx / IN, i = idx - o * IN;
  __bf16* row = wcat + (size_t)o * IN * 9;
  row[i] = (__bf16)base_w[idx];
  float sc = scaler[idx];
  const float* sw = spline_w + (size_t)idx * 8;
#pragma unroll
  for (int c = 0; c < 8; ++c) row[IN + i * 8 + c] = (__bf16)(sw[c] * sc);
}

// ---------------------------------------------------------------------------
// Build KAN activation matrix A[M][IN*9] from h[M][IN]:
//   cols [0,IN)       = silu(x)
//   cols IN + i*8 + c = cubic B-spline basis c of x  (grid: 0.4*(t-3)-1)
// ---------------------------------------------------------------------------
__global__ void build_kan_A(const float* __restrict__ h, __bf16* __restrict__ A,
                            int total, int IN) {
  int idx = blockIdx.x * 256 + threadIdx.x;
  if (idx >= total) return;
  int n = idx / IN, i = idx - n * IN;
  float x = h[idx];
  __bf16* row = A + (size_t)n * IN * 9;
  row[i] = (__bf16)(x / (1.f + __expf(-x)));  // silu
  float b[11];
#pragma unroll
  for (int t = 0; t < 11; ++t) {
    float t0 = 0.4f * (t - 3) - 1.0f;
    float t1 = 0.4f * (t - 2) - 1.0f;
    b[t] = (x >= t0 && x < t1) ? 1.f : 0.f;
  }
#pragma unroll
  for (int j = 1; j <= 3; ++j) {
#pragma unroll
    for (int t = 0; t < 10; ++t) {
      if (t < 11 - j) {
        float g_t   = 0.4f * (t - 3) - 1.0f;
        float g_t1  = 0.4f * (t - 2) - 1.0f;
        float g_tj  = 0.4f * (t + j - 3) - 1.0f;
        float g_tj1 = 0.4f * (t + j - 2) - 1.0f;
        b[t] = (x - g_t) / (g_tj - g_t) * b[t] + (g_tj1 - x) / (g_tj1 - g_t1) * b[t + 1];
      }
    }
  }
#pragma unroll
  for (int c = 0; c < 8; ++c) row[IN + i * 8 + c] = (__bf16)b[c];
}

// ---------------------------------------------------------------------------
// LayerNorm over D=768, one token per 256-thread block. Optional f32/bf16 out.
// ---------------------------------------------------------------------------
__global__ void ln_kernel(const float* __restrict__ xin, const float* __restrict__ g,
                          const float* __restrict__ bta, float* __restrict__ outf,
                          __bf16* __restrict__ outb) {
  const int row = blockIdx.x, t = threadIdx.x;
  const float* xr = xin + (size_t)row * 768;
  float v[3], s = 0.f, s2 = 0.f;
#pragma unroll
  for (int e = 0; e < 3; ++e) { v[e] = xr[t + 256 * e]; s += v[e]; s2 += v[e] * v[e]; }
  __shared__ float r1[256], r2[256];
  r1[t] = s; r2[t] = s2; __syncthreads();
  for (int st = 128; st > 0; st >>= 1) {
    if (t < st) { r1[t] += r1[t + st]; r2[t] += r2[t + st]; }
    __syncthreads();
  }
  float mean = r1[0] * (1.f / 768.f);
  float var  = r2[0] * (1.f / 768.f) - mean * mean;
  float rstd = rsqrtf(var + 1e-5f);
#pragma unroll
  for (int e = 0; e < 3; ++e) {
    int c = t + 256 * e;
    float y = (v[e] - mean) * rstd * g[c] + bta[c];
    if (outf) outf[(size_t)row * 768 + c] = y;
    if (outb) outb[(size_t)row * 768 + c] = (__bf16)y;
  }
}

// ---------------------------------------------------------------------------
// Tiled bf16 GEMM:  out[M][Nout] = A[M][K] * W[Nout][K]^T   (+ epilogue)
// 256 thr = 8 waves (4x2); block tile 256(M)x128(N); wave tile 64x64.
// Double-buffered LDS staged via async global->LDS copies (ASYNCcnt).
// MODE 0: f32 store. MODE 1: +bias, scatter bf16 Q/K/V (b,h,s,d).
// MODE 2: f32 store of resid + acc.
// ---------------------------------------------------------------------------
template <int MODE>
__global__ void __launch_bounds__(256)
gemm_bf16_kernel(const __bf16* __restrict__ A, const __bf16* __restrict__ W,
                 int M, int N, int K,
                 float* __restrict__ outf, const float* __restrict__ resid,
                 const float* __restrict__ bq, const float* __restrict__ bk,
                 const float* __restrict__ bv,
                 __bf16* __restrict__ oq, __bf16* __restrict__ ok, __bf16* __restrict__ ov) {
  __shared__ __bf16 As[2][256][40];
  __shared__ __bf16 Bs[2][128][40];
  const int tid = threadIdx.x;
  const int lane = tid & 31, wave = tid >> 5;
  const int wmi = wave >> 1, wni = wave & 1;   // 4x2 wave grid
  const int m0 = blockIdx.y * 256, n0 = blockIdx.x * 128;

  v8f acc[4][4];
#pragma unroll
  for (int i = 0; i < 4; ++i)
#pragma unroll
    for (int j = 0; j < 4; ++j) acc[i][j] = vzero8();

  // Stage one 32-wide K slice of A(256 rows) and W(128 rows): 6 b128 per thread.
  auto issue_tile = [&](int buf, int k0) {
#pragma unroll
    for (int i = 0; i < 4; ++i) {
      int s = tid + 256 * i;
      int row = s >> 2, seg = s & 3;
      async_cp16(A + (size_t)(m0 + row) * K + k0 + seg * 8, &As[buf][row][seg * 8]);
    }
#pragma unroll
    for (int i = 0; i < 2; ++i) {
      int s = tid + 256 * i;
      int row = s >> 2, seg = s & 3;
      async_cp16(W + (size_t)(n0 + row) * K + k0 + seg * 8, &Bs[buf][row][seg * 8]);
    }
  };

  const int nkt = K >> 5;
  issue_tile(0, 0);
  for (int kt = 0; kt < nkt; ++kt) {
    const int cur = kt & 1;
    if (kt + 1 < nkt) {
      issue_tile(cur ^ 1, (kt + 1) << 5);  // prefetch next slice into other buffer
      WAIT_ASYNCCNT(6);                    // in-order: current slice has landed
    } else {
      WAIT_ASYNCCNT(0);
    }
    __syncthreads();

    v16bf bfr[4];
#pragma unroll
    for (int j = 0; j < 4; ++j)
      bfr[j] = load_b_frag(&Bs[cur][0][0], 40, wni * 64 + j * 16, 0, lane);
#pragma unroll
    for (int i = 0; i < 4; ++i) {
      v16bf af = load_a_frag(&As[cur][0][0], 40, wmi * 64 + i * 16, 0, lane);
#pragma unroll
      for (int j = 0; j < 4; ++j)
        acc[i][j] = __builtin_amdgcn_wmma_f32_16x16x32_bf16(false, af, false, bfr[j],
                                                            (short)0, acc[i][j], false, false);
    }
    __syncthreads();  // protect buffer reuse (WAR vs next issue)
  }

  const int r = lane & 15, hf = lane >> 4;
#pragma unroll
  for (int i = 0; i < 4; ++i)
#pragma unroll
    for (int j = 0; j < 4; ++j)
#pragma unroll
      for (int v = 0; v < 8; ++v) {
        int gm = m0 + wmi * 64 + i * 16 + v + hf * 8;  // C layout: row = v + 8*half
        int gn = n0 + wni * 64 + j * 16 + r;           //           col = lane&15
        float val = acc[i][j][v];
        if (MODE == 0) {
          outf[(size_t)gm * N + gn] = val;
        } else if (MODE == 1) {
          int which = gn / 768, rem = gn - which * 768;
          val += (which == 0 ? bq : which == 1 ? bk : bv)[rem];
          int hh = rem >> 6, dk = rem & 63;
          int bb = gm >> 11, ss = gm & 2047;
          __bf16* dst = which == 0 ? oq : which == 1 ? ok : ov;
          dst[(((size_t)(bb * 12 + hh)) * 2048 + ss) * 64 + dk] = (__bf16)val;
        } else {
          size_t idx = (size_t)gm * N + gn;
          outf[idx] = resid[idx] + val;
        }
      }
}

// ---------------------------------------------------------------------------
// Flash attention (B=2,H=12,S=2048,DK=64). grid=(B*H, S/128), 256 thr = 8 waves,
// 16 query rows per wave. Writes x1 = x + attn (fp32, [N][768]).
// ---------------------------------------------------------------------------
__global__ void __launch_bounds__(256)
attn_kernel(const float* __restrict__ x, const __bf16* __restrict__ qbf,
            const __bf16* __restrict__ kbf, const __bf16* __restrict__ vbf,
            float* __restrict__ x1) {
  __shared__ __bf16 Ks[64][72];     // [key][dk]
  __shared__ __bf16 Vt[64][72];     // [dk][key] (transposed)
  __shared__ __bf16 Ps[8][16][72];  // wave-private P tiles [row][key]
  const int tid = threadIdx.x, lane = tid & 31, w = tid >> 5;
  const int bh = blockIdx.x;
  const int q0 = blockIdx.y * 128 + w * 16;
  const int r = lane & 15, hf = lane >> 4;

  const __bf16* qbase = qbf + ((size_t)bh * 2048 + q0) * 64;
  v16bf qa[2];
  qa[0] = load_a_frag(qbase, 64, 0, 0, lane);
  qa[1] = load_a_frag(qbase, 64, 0, 32, lane);

  v8f o[4];
#pragma unroll
  for (int dt = 0; dt < 4; ++dt) o[dt] = vzero8();
  float mrow[8], srow[8];
#pragma unroll
  for (int v = 0; v < 8; ++v) { mrow[v] = -3.0e38f; srow[v] = 0.f; }

  const int lkey = tid >> 2, lseg = tid & 3;
  for (int j0 = 0; j0 < 2048; j0 += 64) {
    __syncthreads();
    const __bf16* ksrc = kbf + ((size_t)bh * 2048 + j0 + lkey) * 64 + lseg * 16;
    *(v8bf*)&Ks[lkey][lseg * 16]     = *(const v8bf*)ksrc;
    *(v8bf*)&Ks[lkey][lseg * 16 + 8] = *(const v8bf*)(ksrc + 8);
    const __bf16* vsrc = vbf + ((size_t)bh * 2048 + j0 + lkey) * 64 + lseg * 16;
    v8bf vv0 = *(const v8bf*)vsrc, vv1 = *(const v8bf*)(vsrc + 8);
#pragma unroll
    for (int e = 0; e < 8; ++e) {
      Vt[lseg * 16 + e][lkey]     = vv0[e];
      Vt[lseg * 16 + 8 + e][lkey] = vv1[e];
    }
    __syncthreads();

    // scores S = (Q K^T) * 1/sqrt(64)
    v8f s[4];
#pragma unroll
    for (int jt = 0; jt < 4; ++jt) {
      s[jt] = vzero8();
#pragma unroll
      for (int ks = 0; ks < 2; ++ks) {
        v16bf bfr = load_b_frag(&Ks[0][0], 72, jt * 16, ks * 32, lane);
        s[jt] = __builtin_amdgcn_wmma_f32_16x16x32_bf16(false, qa[ks], false, bfr,
                                                        (short)0, s[jt], false, false);
      }
    }

    // online softmax (row = v + 8*half; cols spread over 16 lanes)
#pragma unroll
    for (int v = 0; v < 8; ++v) {
      float mloc = -3.0e38f;
#pragma unroll
      for (int jt = 0; jt < 4; ++jt) { s[jt][v] *= 0.125f; mloc = fmaxf(mloc, s[jt][v]); }
#pragma unroll
      for (int msk = 1; msk <= 8; msk <<= 1) mloc = fmaxf(mloc, __shfl_xor(mloc, msk, 32));
      float mnew = fmaxf(mrow[v], mloc);
      float corr = __expf(mrow[v] - mnew);
      float rs = 0.f;
#pragma unroll
      for (int jt = 0; jt < 4; ++jt) {
        float p = __expf(s[jt][v] - mnew);
        s[jt][v] = p; rs += p;
      }
#pragma unroll
      for (int msk = 1; msk <= 8; msk <<= 1) rs += __shfl_xor(rs, msk, 32);
      srow[v] = srow[v] * corr + rs;
      mrow[v] = mnew;
#pragma unroll
      for (int dt = 0; dt < 4; ++dt) o[dt][v] *= corr;
    }

    // C-layout -> A-layout via wave-private LDS (same-wave DS ops in order)
#pragma unroll
    for (int jt = 0; jt < 4; ++jt)
#pragma unroll
      for (int v = 0; v < 8; ++v)
        Ps[w][v + hf * 8][jt * 16 + r] = (__bf16)s[jt][v];

    // O += P * V
#pragma unroll
    for (int ks = 0; ks < 2; ++ks) {
      v16bf pa = load_a_frag(&Ps[w][0][0], 72, 0, ks * 32, lane);
#pragma unroll
      for (int dt = 0; dt < 4; ++dt) {
        v16bf vbfr = load_b_frag(&Vt[0][0], 72, dt * 16, ks * 32, lane);
        o[dt] = __builtin_amdgcn_wmma_f32_16x16x32_bf16(false, pa, false, vbfr,
                                                        (short)0, o[dt], false, false);
      }
    }
  }

  const int bb = bh / 12, hh = bh % 12;
#pragma unroll
  for (int dt = 0; dt < 4; ++dt)
#pragma unroll
    for (int v = 0; v < 8; ++v) {
      int n = bb * 2048 + q0 + v + hf * 8;
      size_t idx = (size_t)n * 768 + hh * 64 + dt * 16 + r;
      x1[idx] = x[idx] + o[dt][v] / srow[v];
    }
}

// ---------------------------------------------------------------------------
// Host-side orchestration
// ---------------------------------------------------------------------------
extern "C" void kernel_launch(void* const* d_in, const int* in_sizes, int n_in,
                              void* d_out, int out_size, void* d_ws, size_t ws_size,
                              hipStream_t stream) {
  const float* x         = (const float*)d_in[0];
  const float* ln1_g     = (const float*)d_in[2];
  const float* ln1_b     = (const float*)d_in[3];
  const float* wq        = (const float*)d_in[4];
  const float* bq        = (const float*)d_in[5];
  const float* wk        = (const float*)d_in[6];
  const float* bk        = (const float*)d_in[7];
  const float* wv        = (const float*)d_in[8];
  const float* bv        = (const float*)d_in[9];
  const float* ln2_g     = (const float*)d_in[10];
  const float* ln2_b     = (const float*)d_in[11];
  const float* base_w1   = (const float*)d_in[12];
  const float* spline_w1 = (const float*)d_in[13];
  const float* scaler1   = (const float*)d_in[14];
  const float* base_w2   = (const float*)d_in[15];
  const float* spline_w2 = (const float*)d_in[16];
  const float* scaler2   = (const float*)d_in[17];
  float* out = (float*)d_out;
  char* ws = (char*)d_ws;

  const int Ntok = 4096, D = 768, F = 3072;
  const int K1 = D * 9, K2 = F * 9;  // 6912, 27648

  size_t off = 0;
  auto take = [&](size_t bytes) -> size_t {
    size_t o = off;
    off += (bytes + 255) & ~(size_t)255;
    return o;
  };
  size_t o_wqkv  = take((size_t)2304 * D * 2);
  size_t o_wcat1 = take((size_t)F * K1 * 2);
  size_t o_wcat2 = take((size_t)D * K2 * 2);
  size_t o_xn    = take((size_t)Ntok * D * 2);
  size_t o_q     = take((size_t)Ntok * 64 * 12 * 2);
  size_t o_k     = take((size_t)Ntok * 64 * 12 * 2);
  size_t o_v     = take((size_t)Ntok * 64 * 12 * 2);
  size_t o_x1    = take((size_t)Ntok * D * 4);
  const size_t fixed = off;

  auto perchunk = [&](int c) -> size_t {
    auto al = [](size_t b) { return (b + 255) & ~(size_t)255; };
    return al((size_t)c * D * 4) + al((size_t)c * K1 * 2) +
           al((size_t)c * F * 4) + al((size_t)c * K2 * 2);
  };
  int CH = 256;  // must stay a multiple of 256 (GEMM M-tile)
  const int cands[4] = {4096, 2048, 1024, 512};
  for (int i = 0; i < 4; ++i)
    if (fixed + perchunk(cands[i]) <= ws_size) { CH = cands[i]; break; }

  size_t o_h  = take((size_t)CH * D * 4);
  size_t o_A1 = take((size_t)CH * K1 * 2);
  size_t o_h1 = take((size_t)CH * F * 4);
  size_t o_A2 = take((size_t)CH * K2 * 2);

  __bf16* wqkv  = (__bf16*)(ws + o_wqkv);
  __bf16* wcat1 = (__bf16*)(ws + o_wcat1);
  __bf16* wcat2 = (__bf16*)(ws + o_wcat2);
  __bf16* xn    = (__bf16*)(ws + o_xn);
  __bf16* qb    = (__bf16*)(ws + o_q);
  __bf16* kb    = (__bf16*)(ws + o_k);
  __bf16* vb    = (__bf16*)(ws + o_v);
  float*  x1    = (float*)(ws + o_x1);
  float*  hbuf  = (float*)(ws + o_h);
  __bf16* A1    = (__bf16*)(ws + o_A1);
  float*  h1    = (float*)(ws + o_h1);
  __bf16* A2    = (__bf16*)(ws + o_A2);

  // ---- weight prep (bf16) ----
  int nw = D * D;
  cvt_f32_bf16<<<(nw + 255) / 256, 256, 0, stream>>>(wq, wqkv, nw);
  cvt_f32_bf16<<<(nw + 255) / 256, 256, 0, stream>>>(wk, wqkv + (size_t)nw, nw);
  cvt_f32_bf16<<<(nw + 255) / 256, 256, 0, stream>>>(wv, wqkv + (size_t)2 * nw, nw);
  build_wcat<<<(F * D + 255) / 256, 256, 0, stream>>>(base_w1, spline_w1, scaler1, wcat1, F * D, D);
  build_wcat<<<(D * F + 255) / 256, 256, 0, stream>>>(base_w2, spline_w2, scaler2, wcat2, D * F, F);

  // ---- LN1 -> bf16, fused QKV GEMM, flash attention (x1 = x + attn) ----
  ln_kernel<<<Ntok, 256, 0, stream>>>(x, ln1_g, ln1_b, nullptr, xn);
  gemm_bf16_kernel<1><<<dim3(2304 / 128, Ntok / 256), 256, 0, stream>>>(
      xn, wqkv, Ntok, 2304, D, nullptr, nullptr, bq, bk, bv, qb, kb, vb);
  attn_kernel<<<dim3(24, 16), 256, 0, stream>>>(x, qb, kb, vb, x1);

  // ---- token-chunked KAN FFN: out = x1 + KAN2(KAN1(LN2(x1))) ----
  for (int c0 = 0; c0 < Ntok; c0 += CH) {
    const float* x1c = x1 + (size_t)c0 * D;
    ln_kernel<<<CH, 256, 0, stream>>>(x1c, ln2_g, ln2_b, hbuf, nullptr);
    build_kan_A<<<(CH * D + 255) / 256, 256, 0, stream>>>(hbuf, A1, CH * D, D);
    gemm_bf16_kernel<0><<<dim3(F / 128, CH / 256), 256, 0, stream>>>(
        A1, wcat1, CH, F, K1, h1, nullptr, nullptr, nullptr, nullptr,
        nullptr, nullptr, nullptr);
    build_kan_A<<<(CH * F + 255) / 256, 256, 0, stream>>>(h1, A2, CH * F, F);
    gemm_bf16_kernel<2><<<dim3(D / 128, CH / 256), 256, 0, stream>>>(
        A2, wcat2, CH, D, K2, out + (size_t)c0 * D, x1c, nullptr, nullptr, nullptr,
        nullptr, nullptr, nullptr);
  }
}
